// FuzzyPatternTM_STE_30227979829789
// MI455X (gfx1250) — matrix operand order repaired and checked by reading, not compiled, not run on hardware
//
#include <hip/hip_runtime.h>
#include <hip/hip_fp16.h>

// Problem constants (match reference)
#define BDIM 2048   // batch
#define FDIM 3072   // features
#define CDIM 4096   // clauses
#define KCLS 100    // classes
#define HALFC 2048  // CDIM/2
#define K2F  6144   // 2*FDIM (GEMM K dimension)

#define KB   32     // K per WMMA stage
#define LDA  40     // padded LDS row stride in f16 (32 + 8 pad)

typedef __attribute__((ext_vector_type(16))) _Float16 v16h;
typedef __attribute__((ext_vector_type(8)))  _Float16 v8h;
typedef __attribute__((ext_vector_type(8)))  float    v8f;
typedef __attribute__((ext_vector_type(4)))  int      v4i;

union V16U { v16h v; v8h h[2]; };

#if __has_builtin(__builtin_amdgcn_global_load_async_to_lds_b128) && \
    __has_builtin(__builtin_amdgcn_s_wait_asynccnt)
#define USE_ASYNC 1
// Builtin signature (from hipcc diagnostic): param0 = int4 in AS1 (global),
// param1 = int4 in AS3 (LDS), then imm offset, imm cpol.
typedef __attribute__((address_space(1))) v4i GV4;   // global int4
typedef __attribute__((address_space(3))) v4i LV4;   // LDS int4
#else
#define USE_ASYNC 0
#endif

// ---------------------------------------------------------------------------
// Kernel 1: W_total (f16) = sigmoid(ta banks), laid out [CDIM][K2F] row-major.
// Rows 0..HALFC-1 = [p_pos | p_pos_inv], rows HALFC.. = [p_neg | p_neg_inv].
// ---------------------------------------------------------------------------
__global__ __launch_bounds__(256) void prep_w(const float* __restrict__ ta_pos,
                                              const float* __restrict__ ta_neg,
                                              const float* __restrict__ ta_pos_inv,
                                              const float* __restrict__ ta_neg_inv,
                                              _Float16* __restrict__ Wh) {
    long idx = ((long)blockIdx.x * 256 + threadIdx.x) * 8;   // 8 elems/thread
    int j = (int)(idx % K2F);
    int c = (int)(idx / K2F);
    int cc = c & (HALFC - 1);
    int jj = (j < FDIM) ? j : (j - FDIM);
    const float* src;
    if (c < HALFC) src = (j < FDIM) ? ta_pos : ta_pos_inv;
    else           src = (j < FDIM) ? ta_neg : ta_neg_inv;
    const float* p = src + (size_t)cc * FDIM + jj;
    _Float16* o = Wh + idx;
#pragma unroll
    for (int i = 0; i < 8; ++i) {
        float s = 1.0f / (1.0f + __expf(-p[i]));   // sigmoid in [0,1], clip is no-op
        o[i] = (_Float16)s;
    }
}

// ---------------------------------------------------------------------------
// Kernel 2: X_combined (f16) = [1-x | x], [BDIM][K2F] row-major.
// ---------------------------------------------------------------------------
__global__ __launch_bounds__(256) void prep_x(const float* __restrict__ x,
                                              _Float16* __restrict__ Xh) {
    long idx = ((long)blockIdx.x * 256 + threadIdx.x) * 8;
    int j = (int)(idx % K2F);
    int b = (int)(idx / K2F);
    _Float16* o = Xh + idx;
    if (j < FDIM) {
        const float* p = x + (size_t)b * FDIM + j;
#pragma unroll
        for (int i = 0; i < 8; ++i) o[i] = (_Float16)(1.0f - p[i]);
    } else {
        const float* p = x + (size_t)b * FDIM + (j - FDIM);
#pragma unroll
        for (int i = 0; i < 8; ++i) o[i] = (_Float16)p[i];
    }
}

// ---------------------------------------------------------------------------
// Kernel 3: scores = X @ W^T  (WMMA f16 -> f32 acc), fused epilogue:
//           clause = exp(-clip(scores * 4/F, 0, 10)) written to d_out slot.
// Block: 256 threads (8 waves) -> 128x128 output tile.
// Wave (wm in 0..3, wn in 0..1) -> 32(M) x 64(N) sub-block = 2x4 WMMA accs.
// Double-buffered LDS; async global->LDS (ASYNCcnt) prefetches one full
// K-stage ahead of the WMMA pipeline.
// ---------------------------------------------------------------------------
__global__ __launch_bounds__(256) void gemm_wmma(const _Float16* __restrict__ Xh,
                                                 const _Float16* __restrict__ Wh,
                                                 float* __restrict__ clause) {
    __shared__ __align__(16) _Float16 ldsA[2][128 * LDA];
    __shared__ __align__(16) _Float16 ldsB[2][128 * LDA];

    const int tid  = threadIdx.x;
    const int lane = tid & 31;
    const int wave = tid >> 5;        // 0..7
    const int wm   = wave & 3;        // M sub-tile (32 rows each)
    const int wn   = wave >> 2;       // N sub-tile (64 cols each)
    const int hi   = lane >> 4;       // half-wave select
    const int lm   = lane & 15;

    const int blockN = blockIdx.x * 128;   // over CDIM
    const int blockM = blockIdx.y * 128;   // over BDIM

    v8f acc[2][4] = {};

    // staging assignment: 256 threads cover 64 rows x (4 x 8 f16) per pass
    const int sr = tid >> 2;          // 0..63
    const int sc = (tid & 3) * 8;     // 0,8,16,24

    const _Float16* gA0 = Xh + (size_t)(blockM + sr)      * K2F + sc;
    const _Float16* gA1 = Xh + (size_t)(blockM + sr + 64) * K2F + sc;
    const _Float16* gB0 = Wh + (size_t)(blockN + sr)      * K2F + sc;
    const _Float16* gB1 = Wh + (size_t)(blockN + sr + 64) * K2F + sc;
    const int lo0 = sr * LDA + sc;
    const int lo1 = (sr + 64) * LDA + sc;

    // stage one 128x32 A tile + 128x32 B tile (4 x b128 per thread)
    auto stage = [&](int buf, int k0) {
#if USE_ASYNC
        __builtin_amdgcn_global_load_async_to_lds_b128((GV4*)(gA0 + k0), (LV4*)&ldsA[buf][lo0], 0, 0);
        __builtin_amdgcn_global_load_async_to_lds_b128((GV4*)(gA1 + k0), (LV4*)&ldsA[buf][lo1], 0, 0);
        __builtin_amdgcn_global_load_async_to_lds_b128((GV4*)(gB0 + k0), (LV4*)&ldsB[buf][lo0], 0, 0);
        __builtin_amdgcn_global_load_async_to_lds_b128((GV4*)(gB1 + k0), (LV4*)&ldsB[buf][lo1], 0, 0);
#else
        *(v8h*)&ldsA[buf][lo0] = *(const v8h*)(gA0 + k0);
        *(v8h*)&ldsA[buf][lo1] = *(const v8h*)(gA1 + k0);
        *(v8h*)&ldsB[buf][lo0] = *(const v8h*)(gB0 + k0);
        *(v8h*)&ldsB[buf][lo1] = *(const v8h*)(gB1 + k0);
        __builtin_prefetch(gA0 + k0 + KB, 0, 1);
        __builtin_prefetch(gB0 + k0 + KB, 0, 1);
#endif
    };

    stage(0, 0);                        // prologue: fill buffer 0

    for (int k0 = 0; k0 < K2F; k0 += KB) {
        const int cur = (k0 / KB) & 1;
        if (k0 + KB < K2F) {            // uniform: prefetch next stage into other buffer
            stage(cur ^ 1, k0 + KB);
#if USE_ASYNC
            __builtin_amdgcn_s_wait_asynccnt(4);   // current stage's 4 xfers done
#endif
        } else {
#if USE_ASYNC
            __builtin_amdgcn_s_wait_asynccnt(0);
#endif
        }
        __syncthreads();                // all waves' LDS writes for `cur` visible

        // Load fragments per CDNA5 VGPR layouts.
        // A (16x32 f16): lanes 0-15 row=lm K0-7 (V0-3) + K16-23 (V4-7);
        //                lanes 16-31 same row, K8-15 + K24-31.
        V16U a[2], b[4];
#pragma unroll
        for (int mi = 0; mi < 2; ++mi) {
            int row = wm * 32 + mi * 16 + lm;
            a[mi].h[0] = *(const v8h*)&ldsA[cur][row * LDA + hi * 8];
            a[mi].h[1] = *(const v8h*)&ldsA[cur][row * LDA + 16 + hi * 8];
        }
        // B (32x16 f16): lane holds one column (= one W row); lanes 0-15 K0-7 + K16-23,
        // lanes 16-31 K8-15 + K24-31 (same K order as A).
#pragma unroll
        for (int ni = 0; ni < 4; ++ni) {
            int col = wn * 64 + ni * 16 + lm;
            b[ni].h[0] = *(const v8h*)&ldsB[cur][col * LDA + hi * 8];
            b[ni].h[1] = *(const v8h*)&ldsB[cur][col * LDA + 16 + hi * 8];
        }
#pragma unroll
        for (int mi = 0; mi < 2; ++mi)
#pragma unroll
            for (int ni = 0; ni < 4; ++ni)
                acc[mi][ni] = __builtin_amdgcn_wmma_f32_16x16x32_f16(
                    false, a[mi].v, false, b[ni].v,
                    (short)0, acc[mi][ni], false, false);

        __syncthreads();                // reads of `cur` done before it is re-staged
    }

    // Epilogue: D layout -> VGPR r holds M=r (lanes 0-15) / M=r+8 (lanes 16-31),
    // N = lane%16.
    const float scale = 4.0f / (float)FDIM;
#pragma unroll
    for (int mi = 0; mi < 2; ++mi) {
#pragma unroll
        for (int ni = 0; ni < 4; ++ni) {
            int n = blockN + wn * 64 + ni * 16 + lm;
#pragma unroll
            for (int r = 0; r < 8; ++r) {
                int m = blockM + wm * 32 + mi * 16 + hi * 8 + r;
                float s = acc[mi][ni][r] * scale;
                s = fminf(fmaxf(s, 0.0f), 10.0f);
                clause[(size_t)m * CDIM + n] = __expf(-s);
            }
        }
    }
}

// ---------------------------------------------------------------------------
// Kernel 4: logits = (clause + bias) @ voting   [2048 x 100], tiny GEMM.
// One block per batch row; clause row staged through LDS.
// ---------------------------------------------------------------------------
__global__ __launch_bounds__(128) void logits_kernel(const float* __restrict__ clause,
                                                     const float* __restrict__ bias,
                                                     const float* __restrict__ voting,
                                                     float* __restrict__ out) {
    __shared__ float srow[512];
    const int b = blockIdx.x;
    const int k = threadIdx.x;
    float acc = 0.0f;
    for (int c0 = 0; c0 < CDIM; c0 += 512) {
        __syncthreads();
        for (int i = threadIdx.x; i < 512; i += 128)
            srow[i] = clause[(size_t)b * CDIM + c0 + i] + bias[c0 + i];
        __syncthreads();
        if (k < KCLS) {
            for (int c = 0; c < 512; ++c)
                acc = fmaf(srow[c], voting[(size_t)(c0 + c) * KCLS + k], acc);
        }
    }
    if (k < KCLS) out[(size_t)b * KCLS + k] = acc;
}

// ---------------------------------------------------------------------------
extern "C" void kernel_launch(void* const* d_in, const int* in_sizes, int n_in,
                              void* d_out, int out_size, void* d_ws, size_t ws_size,
                              hipStream_t stream) {
    const float* x           = (const float*)d_in[0];
    const float* ta_pos      = (const float*)d_in[1];
    const float* ta_neg      = (const float*)d_in[2];
    const float* ta_pos_inv  = (const float*)d_in[3];
    const float* ta_neg_inv  = (const float*)d_in[4];
    const float* clause_bias = (const float*)d_in[5];
    const float* voting      = (const float*)d_in[6];

    // Workspace layout: W f16 [CDIM*K2F] then X f16 [BDIM*K2F]  (~72 MB)
    _Float16* Wh = (_Float16*)d_ws;
    _Float16* Xh = (_Float16*)((char*)d_ws + (size_t)CDIM * K2F * sizeof(_Float16));

    // Output layout: logits [BDIM*KCLS] then clause_outputs [BDIM*CDIM]
    float* logits = (float*)d_out;
    float* clause = logits + (size_t)BDIM * KCLS;

    {
        long total = (long)CDIM * K2F;                 // 25,165,824
        int blocks = (int)(total / (256 * 8));         // 12288
        prep_w<<<blocks, 256, 0, stream>>>(ta_pos, ta_neg, ta_pos_inv, ta_neg_inv, Wh);
    }
    {
        long total = (long)BDIM * K2F;                 // 12,582,912
        int blocks = (int)(total / (256 * 8));         // 6144
        prep_x<<<blocks, 256, 0, stream>>>(x, Xh);
    }
    {
        dim3 grid(CDIM / 128, BDIM / 128);             // 32 x 16
        gemm_wmma<<<grid, 256, 0, stream>>>(Xh, Wh, clause);
    }
    logits_kernel<<<BDIM, 128, 0, stream>>>(clause, clause_bias, voting, logits);
}
